// TinySeq2Seq_79654463471790
// MI455X (gfx1250) — matrix operand order repaired and checked by reading, not compiled, not run on hardware
//
#include <hip/hip_runtime.h>
#include <math.h>

// ---------------- constants ----------------
#define BB   8
#define SS   1024
#define LL   1025          // 1 cls + 1024 tokens
#define LP   1056          // L padded to multiple of 32
#define DM   512
#define TD   1536          // 3*D
#define HH   8
#define HD   64
#define NLAY 6
#define FFD  2048
#define MAXD 64
#define NOUT 80            // OUTLEN*NC
#define MP   (BB*LP)       // 8448 padded token rows

typedef __attribute__((ext_vector_type(16))) __bf16 bf16x16;
typedef __attribute__((ext_vector_type(8)))  float  floatx8;

union Frag {
    bf16x16 v;
    uint4 q2[2];
    unsigned short s[16];
};

__device__ inline floatx8 zero8() {
    floatx8 z;
#pragma unroll
    for (int i = 0; i < 8; ++i) z[i] = 0.0f;
    return z;
}

__device__ inline unsigned short f2bf(float f) {
    unsigned int u = __float_as_uint(f);
    unsigned int r = (u + 0x7FFFu + ((u >> 16) & 1u)) >> 16;
    return (unsigned short)r;
}

__device__ inline float redmax16(float v) {
#pragma unroll
    for (int m = 1; m < 16; m <<= 1) v = fmaxf(v, __shfl_xor(v, m, 32));
    return v;
}
__device__ inline float redsum16(float v) {
#pragma unroll
    for (int m = 1; m < 16; m <<= 1) v += __shfl_xor(v, m, 32);
    return v;
}
__device__ inline float redsum32(float v) {
#pragma unroll
    for (int m = 1; m < 32; m <<= 1) v += __shfl_xor(v, m, 32);
    return v;
}

__device__ inline float gelu_exact(float x) {
    return 0.5f * x * (1.0f + erff(x * 0.70710678118654752f));
}

// ---------------- fp32 -> bf16 convert ----------------
__global__ __launch_bounds__(256) void cvt_kernel(const float* __restrict__ in,
                                                  unsigned short* __restrict__ out, int n) {
    int i = (blockIdx.x * 256 + threadIdx.x) * 4;
    if (i + 3 < n) {
        float4 v = *(const float4*)(in + i);
        ushort4 o = make_ushort4(f2bf(v.x), f2bf(v.y), f2bf(v.z), f2bf(v.w));
        *(ushort4*)(out + i) = o;
    } else {
        for (; i < n; ++i) out[i] = f2bf(in[i]);
    }
}

// ---------------- embedding (+cls, +pad-zero) ----------------
__global__ __launch_bounds__(128) void embed_kernel(const int* __restrict__ src,
                                                    const float* __restrict__ emb,
                                                    const float* __restrict__ cls,
                                                    float* __restrict__ xF,
                                                    unsigned short* __restrict__ xB) {
    int row = blockIdx.x;
    int b = row / LP, l = row % LP;
    int d0 = threadIdx.x * 4;
    float4 v;
    if (l == 0) {
        v = *(const float4*)(cls + d0);
    } else if (l <= SS) {
        int tok = src[b * SS + (l - 1)];
        v = *(const float4*)(emb + (size_t)tok * DM + d0);
    } else {
        v = make_float4(0.f, 0.f, 0.f, 0.f);
    }
    *(float4*)(xF + (size_t)row * DM + d0) = v;
    *(ushort4*)(xB + (size_t)row * DM + d0) =
        make_ushort4(f2bf(v.x), f2bf(v.y), f2bf(v.z), f2bf(v.w));
}

// ---------------- generic bf16 WMMA GEMM: C = A(MxK) * W(NxK)^T + bias ----------------
// block = 256 threads (8 waves), tile 128(M) x 64(N), each wave 16x64
// Software-pipelined: next K-step fragments are fetched while current WMMAs issue.
__global__ __launch_bounds__(256) void gemm_bf16_kernel(
    const unsigned short* __restrict__ A, const unsigned short* __restrict__ W,
    const float* __restrict__ bias, float* __restrict__ outF,
    unsigned short* __restrict__ outB, int M, int N, int K, int act) {
    const int lane = threadIdx.x & 31;
    const int wave = threadIdx.x >> 5;
    const int half = lane >> 4;
    const int l16 = lane & 15;
    const int m0 = blockIdx.y * 128 + wave * 16;
    const int n0 = blockIdx.x * 64;

    floatx8 acc[4];
#pragma unroll
    for (int t = 0; t < 4; ++t) acc[t] = zero8();

    const unsigned short* Arow = A + (size_t)(m0 + l16) * K;
    const unsigned short* W0 = W + (size_t)(n0 + l16) * K;

    // prologue: fragments for k0 = 0
    Frag a_c, b_c[4];
    a_c.q2[0] = *(const uint4*)(Arow + half * 16);
    a_c.q2[1] = *(const uint4*)(Arow + half * 16 + 8);
#pragma unroll
    for (int t = 0; t < 4; ++t) {
        const unsigned short* Wr = W0 + (size_t)t * 16 * K + half * 16;
        b_c[t].q2[0] = *(const uint4*)(Wr);
        b_c[t].q2[1] = *(const uint4*)(Wr + 8);
    }

    for (int k0 = 0; k0 < K; k0 += 32) {
        // branch-free next-step index (last iteration re-fetches k=0, discarded)
        int k1 = (k0 + 32 < K) ? (k0 + 32) : 0;
        __builtin_prefetch(Arow + k0 + 128, 0, 3);
        __builtin_prefetch(W0 + k0 + 128, 0, 3);
        Frag a_n, b_n[4];
        a_n.q2[0] = *(const uint4*)(Arow + k1 + half * 16);
        a_n.q2[1] = *(const uint4*)(Arow + k1 + half * 16 + 8);
#pragma unroll
        for (int t = 0; t < 4; ++t) {
            const unsigned short* Wr = W0 + (size_t)t * 16 * K + k1 + half * 16;
            b_n[t].q2[0] = *(const uint4*)(Wr);
            b_n[t].q2[1] = *(const uint4*)(Wr + 8);
        }
#pragma unroll
        for (int t = 0; t < 4; ++t) {
            acc[t] = __builtin_amdgcn_wmma_f32_16x16x32_bf16(
                false, a_c.v, false, b_c[t].v, (short)0, acc[t], false, false);
        }
        a_c = a_n;
#pragma unroll
        for (int t = 0; t < 4; ++t) b_c[t] = b_n[t];
    }

#pragma unroll
    for (int t = 0; t < 4; ++t) {
        int n = n0 + t * 16 + l16;
        float bv = bias ? bias[n] : 0.0f;
#pragma unroll
        for (int r = 0; r < 8; ++r) {
            int m = m0 + r + half * 8;
            float v = acc[t][r] + bv;
            if (act == 1) v = gelu_exact(v);
            size_t idx = (size_t)m * N + n;
            if (outF) outF[idx] = v;
            if (outB) outB[idx] = f2bf(v);
        }
    }
}

// ---------------- flash attention (1 wave / block) ----------------
// grid: (LP/16 qtiles, H, B)
__global__ __launch_bounds__(32) void attn_kernel(const unsigned short* __restrict__ qkv,
                                                  const float* __restrict__ rel, // layer slice (129 x H)
                                                  const unsigned char* __restrict__ pad,
                                                  unsigned short* __restrict__ o_out) {
    const int lane = threadIdx.x;
    const int half = lane >> 4;
    const int l16 = lane & 15;
    const int qt = blockIdx.x, h = blockIdx.y, b = blockIdx.z;

    __shared__ __align__(16) unsigned short Pl[16 * 40];
    __shared__ __align__(16) unsigned short Vl[32 * 72];

    // Q A-fragments (16 queries x 64 hd), two K-steps of 32
    Frag qf[2];
    {
        const unsigned short* qb =
            qkv + (size_t)(b * LP + qt * 16 + l16) * TD + (size_t)h * HD;
#pragma unroll
        for (int ks = 0; ks < 2; ++ks) {
            qf[ks].q2[0] = *(const uint4*)(qb + ks * 32 + half * 16);
            qf[ks].q2[1] = *(const uint4*)(qb + ks * 32 + half * 16 + 8);
        }
    }

    floatx8 o[4];
#pragma unroll
    for (int t = 0; t < 4; ++t) o[t] = zero8();
    float mrun[8], lrun[8];
#pragma unroll
    for (int r = 0; r < 8; ++r) { mrun[r] = -3.0e38f; lrun[r] = 0.0f; }

    for (int c = 0; c < LP / 32; ++c) {
        // ---- start V row load early (1 key row per lane, coalesced b128) ----
        uint4 vreg[8];
        {
            const unsigned short* vrow =
                qkv + (size_t)(b * LP + c * 32 + lane) * TD + 2 * DM + (size_t)h * HD;
#pragma unroll
            for (int i = 0; i < 8; ++i) vreg[i] = *(const uint4*)(vrow + i * 8);
        }
        // prefetch next chunk's K and V rows
        {
            int cn = (c + 1 < LP / 32) ? (c + 1) : c;
            const unsigned short* nk =
                qkv + (size_t)(b * LP + cn * 32 + lane) * TD + DM + (size_t)h * HD;
            __builtin_prefetch(nk, 0, 3);
            __builtin_prefetch(nk + DM, 0, 3);
        }

        // ---- S = Q * K^T for 32 keys (two 16x16 tiles) ----
        floatx8 s[2];
#pragma unroll
        for (int st = 0; st < 2; ++st) {
            s[st] = zero8();
            const unsigned short* kb =
                qkv + (size_t)(b * LP + c * 32 + st * 16 + l16) * TD + DM + (size_t)h * HD;
#pragma unroll
            for (int ks = 0; ks < 2; ++ks) {
                Frag kf;
                kf.q2[0] = *(const uint4*)(kb + ks * 32 + half * 16);
                kf.q2[1] = *(const uint4*)(kb + ks * 32 + half * 16 + 8);
                s[st] = __builtin_amdgcn_wmma_f32_16x16x32_bf16(
                    false, qf[ks].v, false, kf.v, (short)0, s[st], false, false);
            }
        }
        // ---- scale + relative bias + mask (C layout) ----
#pragma unroll
        for (int st = 0; st < 2; ++st) {
#pragma unroll
            for (int r = 0; r < 8; ++r) {
                int qpos = qt * 16 + r + half * 8;
                int kpos = c * 32 + st * 16 + l16;
                float val = s[st][r] * 0.125f;
                int dd = kpos - qpos;
                dd = dd < -MAXD ? -MAXD : (dd > MAXD ? MAXD : dd);
                dd += MAXD;
                val += rel[dd * HH + h];
                bool bad = (kpos >= LL) ||
                           ((kpos >= 1) && (pad[b * SS + (kpos - 1)] != 0));
                s[st][r] = bad ? -1.0e30f : val;
            }
        }
        // ---- online softmax ----
        float alpha[8];
#pragma unroll
        for (int r = 0; r < 8; ++r) {
            float mx = redmax16(fmaxf(s[0][r], s[1][r]));
            float mn = fmaxf(mrun[r], mx);
            alpha[r] = expf(mrun[r] - mn);
            mrun[r] = mn;
        }
#pragma unroll
        for (int st = 0; st < 2; ++st)
#pragma unroll
            for (int r = 0; r < 8; ++r) s[st][r] = expf(s[st][r] - mrun[r]);
#pragma unroll
        for (int r = 0; r < 8; ++r) {
            float rs = redsum16(s[0][r] + s[1][r]);
            lrun[r] = lrun[r] * alpha[r] + rs;
        }
#pragma unroll
        for (int t = 0; t < 4; ++t)
#pragma unroll
            for (int r = 0; r < 8; ++r) o[t][r] *= alpha[r];

        // ---- P (16x32) and V (32x64) -> LDS ----
        __syncthreads();
#pragma unroll
        for (int st = 0; st < 2; ++st)
#pragma unroll
            for (int r = 0; r < 8; ++r)
                Pl[(r + half * 8) * 40 + st * 16 + l16] = f2bf(s[st][r]);
        {
            unsigned short* dst = Vl + lane * 72;
#pragma unroll
            for (int i = 0; i < 8; ++i) *(uint4*)(dst + i * 8) = vreg[i];
        }
        __syncthreads();

        Frag pf;
        pf.q2[0] = *(const uint4*)(Pl + l16 * 40 + half * 16);
        pf.q2[1] = *(const uint4*)(Pl + l16 * 40 + half * 16 + 8);

        // ---- O += P(16x32) * V(32x64), V gathered from LDS ----
#pragma unroll
        for (int t = 0; t < 4; ++t) {
            Frag vf;
#pragma unroll
            for (int e = 0; e < 16; ++e) {
                int kk = half * 16 + e;
                vf.s[e] = Vl[kk * 72 + t * 16 + l16];
            }
            o[t] = __builtin_amdgcn_wmma_f32_16x16x32_bf16(
                false, pf.v, false, vf.v, (short)0, o[t], false, false);
        }
        __syncthreads();
    }

    // ---- normalize and write bf16 ----
#pragma unroll
    for (int t = 0; t < 4; ++t)
#pragma unroll
        for (int r = 0; r < 8; ++r) {
            int qpos = qt * 16 + r + half * 8;
            float val = o[t][r] / lrun[r];
            o_out[(size_t)(b * LP + qpos) * DM + h * HD + t * 16 + l16] = f2bf(val);
        }
}

// ---------------- residual + LayerNorm (1 wave / row) ----------------
__global__ __launch_bounds__(32) void ln_kernel(const float* __restrict__ a,
                                                const float* __restrict__ resid,
                                                const float* __restrict__ g,
                                                const float* __restrict__ be,
                                                float* __restrict__ outF,
                                                unsigned short* __restrict__ outB) {
    const int row = blockIdx.x, lane = threadIdx.x;
    const float* ar = a + (size_t)row * DM + lane * 16;
    const float* rr = resid ? resid + (size_t)row * DM + lane * 16 : nullptr;
    float v[16];
    float s = 0.f, s2 = 0.f;
#pragma unroll
    for (int i = 0; i < 16; ++i) {
        float x = ar[i] + (rr ? rr[i] : 0.f);
        v[i] = x;
        s += x;
        s2 += x * x;
    }
    s = redsum32(s);
    s2 = redsum32(s2);
    float mean = s * (1.0f / DM);
    float var = s2 * (1.0f / DM) - mean * mean;
    float rstd = rsqrtf(var + 1e-5f);
#pragma unroll
    for (int i = 0; i < 16; ++i) {
        int d = lane * 16 + i;
        float y = (v[i] - mean) * rstd * g[d] + be[d];
        size_t idx = (size_t)row * DM + d;
        if (outF) outF[idx] = y;
        if (outB) outB[idx] = f2bf(y);
    }
}

// ---------------- classifier (cls token) ----------------
__global__ __launch_bounds__(32) void cls_kernel(const float* __restrict__ xn,
                                                 const float* __restrict__ w,
                                                 const float* __restrict__ bias,
                                                 float* __restrict__ out) {
    int o = blockIdx.x, b = blockIdx.y, lane = threadIdx.x;
    const float* xr = xn + (size_t)(b * LP) * DM + lane * 16;
    const float* wr = w + (size_t)o * DM + lane * 16;
    float s = 0.f;
#pragma unroll
    for (int i = 0; i < 16; ++i) s += xr[i] * wr[i];
    s = redsum32(s);
    if (lane == 0) out[b * NOUT + o] = s + bias[o];
}

// ---------------- host orchestration ----------------
extern "C" void kernel_launch(void* const* d_in, const int* in_sizes, int n_in,
                              void* d_out, int out_size, void* d_ws, size_t ws_size,
                              hipStream_t stream) {
    const int*           src      = (const int*)d_in[0];
    const unsigned char* pad_mask = (const unsigned char*)d_in[1];
    const float* embed_w = (const float*)d_in[2];
    const float* cls_tok = (const float*)d_in[3];
    const float* qkv_w   = (const float*)d_in[4];
    const float* qkv_b   = (const float*)d_in[5];
    const float* out_w   = (const float*)d_in[6];
    const float* out_b   = (const float*)d_in[7];
    const float* rel_emb = (const float*)d_in[8];
    const float* ln1_g   = (const float*)d_in[9];
    const float* ln1_b   = (const float*)d_in[10];
    const float* w1      = (const float*)d_in[11];
    const float* b1      = (const float*)d_in[12];
    const float* w2      = (const float*)d_in[13];
    const float* b2      = (const float*)d_in[14];
    const float* ln2_g   = (const float*)d_in[15];
    const float* ln2_b   = (const float*)d_in[16];
    const float* norm_g  = (const float*)d_in[17];
    const float* norm_b  = (const float*)d_in[18];
    const float* cls_w   = (const float*)d_in[19];
    const float* cls_b   = (const float*)d_in[20];

    char* base = (char*)d_ws;
    size_t off = 0;
    auto alloc = [&](size_t bytes) -> void* {
        off = (off + 255) & ~(size_t)255;
        void* p = base + off;
        off += bytes;
        return p;
    };

    float*          xF    = (float*)alloc((size_t)MP * DM * 4);
    unsigned short* xB    = (unsigned short*)alloc((size_t)MP * DM * 2);
    float*          x2F   = (float*)alloc((size_t)MP * DM * 4);
    unsigned short* x2B   = (unsigned short*)alloc((size_t)MP * DM * 2);
    float*          tmpF  = (float*)alloc((size_t)MP * DM * 4);
    unsigned short* qkvB  = (unsigned short*)alloc((size_t)MP * TD * 2);
    unsigned short* oB    = (unsigned short*)alloc((size_t)MP * DM * 2);
    unsigned short* hB    = (unsigned short*)alloc((size_t)MP * FFD * 2);
    unsigned short* wqkvB = (unsigned short*)alloc((size_t)NLAY * TD * DM * 2);
    unsigned short* woutB = (unsigned short*)alloc((size_t)NLAY * DM * DM * 2);
    unsigned short* w1B   = (unsigned short*)alloc((size_t)NLAY * FFD * DM * 2);
    unsigned short* w2B   = (unsigned short*)alloc((size_t)NLAY * DM * FFD * 2);
    if (off > ws_size) return;  // workspace too small; nothing safe to do

    auto cvt = [&](const float* s, unsigned short* d, size_t n) {
        cvt_kernel<<<(unsigned)((n / 4 + 255) / 256), 256, 0, stream>>>(s, d, (int)n);
    };
    cvt(qkv_w, wqkvB, (size_t)NLAY * TD * DM);
    cvt(out_w, woutB, (size_t)NLAY * DM * DM);
    cvt(w1, w1B, (size_t)NLAY * FFD * DM);
    cvt(w2, w2B, (size_t)NLAY * DM * FFD);

    embed_kernel<<<MP, 128, 0, stream>>>(src, embed_w, cls_tok, xF, xB);

    const unsigned gy = MP / 128;  // 66
    for (int lyr = 0; lyr < NLAY; ++lyr) {
        // QKV: [MP x 1536] = xB * qkv_w^T + qkv_b  -> bf16
        gemm_bf16_kernel<<<dim3(TD / 64, gy), 256, 0, stream>>>(
            xB, wqkvB + (size_t)lyr * TD * DM, qkv_b + (size_t)lyr * TD,
            nullptr, qkvB, MP, TD, DM, 0);
        // attention -> oB (bf16)
        attn_kernel<<<dim3(LP / 16, HH, BB), 32, 0, stream>>>(
            qkvB, rel_emb + (size_t)lyr * (2 * MAXD + 1) * HH, pad_mask, oB);
        // out-proj -> tmpF (f32)
        gemm_bf16_kernel<<<dim3(DM / 64, gy), 256, 0, stream>>>(
            oB, woutB + (size_t)lyr * DM * DM, out_b + (size_t)lyr * DM,
            tmpF, nullptr, MP, DM, DM, 0);
        // x2 = LN(x + attn_out)
        ln_kernel<<<MP, 32, 0, stream>>>(tmpF, xF, ln1_g + (size_t)lyr * DM,
                                         ln1_b + (size_t)lyr * DM, x2F, x2B);
        // h = gelu(x2 * w1^T + b1) -> bf16
        gemm_bf16_kernel<<<dim3(FFD / 64, gy), 256, 0, stream>>>(
            x2B, w1B + (size_t)lyr * FFD * DM, b1 + (size_t)lyr * FFD,
            nullptr, hB, MP, FFD, DM, 1);
        // ff = h * w2^T + b2 -> tmpF (f32)
        gemm_bf16_kernel<<<dim3(DM / 64, gy), 256, 0, stream>>>(
            hB, w2B + (size_t)lyr * DM * FFD, b2 + (size_t)lyr * DM,
            tmpF, nullptr, MP, DM, FFD, 0);
        // x = LN(x2 + ff)
        ln_kernel<<<MP, 32, 0, stream>>>(tmpF, x2F, ln2_g + (size_t)lyr * DM,
                                         ln2_b + (size_t)lyr * DM, xF, xB);
    }

    // final LN -> tmpF, then classifier on cls rows
    ln_kernel<<<MP, 32, 0, stream>>>(xF, nullptr, norm_g, norm_b, tmpF, nullptr);
    cls_kernel<<<dim3(NOUT, BB), 32, 0, stream>>>(tmpF, cls_w, cls_b, (float*)d_out);
}
